// BitLinear_52905407152468
// MI455X (gfx1250) — 8-bit matrix operand order repaired and checked by reading, compiled, not run on hardware
//
#include <hip/hip_runtime.h>
#include <stdint.h>

typedef __attribute__((ext_vector_type(8))) int v8i;

#define DDIM   4096
#define MTOK   8192
#define ODIM   16384
#define QMAX   127.0f

__global__ void __launch_bounds__(256) quant_x_kernel(const float* __restrict__ x,
                                                      int8_t* __restrict__ xq,
                                                      float* __restrict__ xscale) {
  __shared__ float red[256];
  const int token = blockIdx.x;
  const int tid   = threadIdx.x;

  const float4* xp = (const float4*)(x + (size_t)token * DDIM + tid * 16);
  union { float4 f4[4]; float f[16]; } u;
  u.f4[0] = xp[0]; u.f4[1] = xp[1]; u.f4[2] = xp[2]; u.f4[3] = xp[3];

  float m = 0.0f;
#pragma unroll
  for (int i = 0; i < 16; ++i) m = fmaxf(m, fabsf(u.f[i]));
  red[tid] = m;
  __syncthreads();
#pragma unroll
  for (int s = 128; s > 0; s >>= 1) {
    if (tid < s) red[tid] = fmaxf(red[tid], red[tid + s]);
    __syncthreads();
  }
  const float scale = fmaxf(red[0] / QMAX, 1e-8f);
  if (tid == 0) xscale[token] = scale;
  const float inv = 1.0f / scale;

  union { int8_t b[16]; int4 v; } q;
#pragma unroll
  for (int i = 0; i < 16; ++i) {
    int qi = (int)rintf(u.f[i] * inv);
    qi = qi > 127 ? 127 : (qi < -127 ? -127 : qi);
    q.b[i] = (int8_t)qi;
  }
  *(int4*)(xq + (size_t)token * DDIM + tid * 16) = q.v;
}

__global__ void init_sum_kernel(double* __restrict__ sum) {
  if (threadIdx.x == 0 && blockIdx.x == 0) *sum = 0.0;
}

__global__ void __launch_bounds__(256) wabs_sum_kernel(const float* __restrict__ w,
                                                       double* __restrict__ sum) {
  __shared__ double red[256];
  const size_t n4 = (size_t)ODIM * DDIM / 4;
  const float4* w4 = (const float4*)w;
  float acc = 0.0f;
  for (size_t j = (size_t)blockIdx.x * 256 + threadIdx.x; j < n4;
       j += (size_t)gridDim.x * 256) {
    float4 v = w4[j];
    acc += fabsf(v.x) + fabsf(v.y) + fabsf(v.z) + fabsf(v.w);
  }
  red[threadIdx.x] = (double)acc;
  __syncthreads();
#pragma unroll
  for (int s = 128; s > 0; s >>= 1) {
    if (threadIdx.x < s) red[threadIdx.x] += red[threadIdx.x + s];
    __syncthreads();
  }
  if (threadIdx.x == 0) atomicAdd(sum, red[0]);
}

__global__ void __launch_bounds__(256) quant_w_kernel(const float* __restrict__ w,
                                                      const double* __restrict__ sum,
                                                      int8_t* __restrict__ wq) {
  const float gamma =
      fmaxf((float)(*sum / (double)((size_t)ODIM * DDIM)), 1e-6f);
  const float inv = 1.0f / gamma;
  const int row = blockIdx.x;
  const int tid = threadIdx.x;

  const float4* wp = (const float4*)(w + (size_t)row * DDIM + tid * 16);
  union { float4 f4[4]; float f[16]; } u;
  u.f4[0] = wp[0]; u.f4[1] = wp[1]; u.f4[2] = wp[2]; u.f4[3] = wp[3];

  union { int8_t b[16]; int4 v; } q;
#pragma unroll
  for (int i = 0; i < 16; ++i) {
    float c = fminf(1.0f, fmaxf(-1.0f, u.f[i] * inv));
    q.b[i] = (int8_t)(int)rintf(c);
  }
  *(int4*)(wq + (size_t)row * DDIM + tid * 16) = q.v;
}

__device__ __forceinline__ void async_load16(uint32_t lds, const int8_t* g) {
  asm volatile("global_load_async_to_lds_b128 %0, %1, off"
               :: "v"(lds), "v"(g)
               : "memory");
}
__device__ __forceinline__ void wait_async0() {
  asm volatile("s_wait_asynccnt 0x0" ::: "memory");
}
__device__ __forceinline__ uint32_t lds_addr32(const void* p) {
  return (uint32_t)(uintptr_t)p;
}

__global__ void __launch_bounds__(256) gemm_iu8_kernel(const int8_t* __restrict__ xq,
                                                       const int8_t* __restrict__ wq,
                                                       const float* __restrict__ xscale,
                                                       const double* __restrict__ sum,
                                                       float* __restrict__ out) {
  __shared__ __align__(16) int8_t As[2][128 * 128];
  __shared__ __align__(16) int8_t Bs[2][128 * 128];
  __shared__ float xs[128];

  const int tid  = threadIdx.x;
  const int lane = tid & 31;
  const int wave = tid >> 5;
  const int h    = lane >> 4;
  const int ln   = lane & 15;
  const int wm   = wave & 1;
  const int wn   = wave >> 1;
  const int bm   = blockIdx.y;
  const int bn   = blockIdx.x;

  const float gamma =
      fmaxf((float)(*sum / (double)((size_t)ODIM * DDIM)), 1e-6f);

  if (tid < 128) xs[tid] = xscale[bm * 128 + tid];

  const int rowS = tid >> 3;
  const int kcS  = (tid & 7) * 16;
  const int8_t* Ag = xq + (size_t)(bm * 128 + rowS) * DDIM + kcS;
  const int8_t* Bg = wq + (size_t)(bn * 128 + rowS) * DDIM + kcS;
  const uint32_t ldsA0 = lds_addr32(&As[0][0]) + rowS * 128 + kcS;
  const uint32_t ldsB0 = lds_addr32(&Bs[0][0]) + rowS * 128 + kcS;

  v8i acc[4][2];
#pragma unroll
  for (int mt = 0; mt < 4; ++mt)
#pragma unroll
    for (int nt = 0; nt < 2; ++nt)
#pragma unroll
      for (int i = 0; i < 8; ++i) acc[mt][nt][i] = 0;

  const int NK = DDIM / 128;

#pragma unroll
  for (int i = 0; i < 4; ++i) {
    const size_t rstep = (size_t)(32 * i) * DDIM;
    async_load16(ldsA0 + 32 * i * 128, Ag + rstep);
    async_load16(ldsB0 + 32 * i * 128, Bg + rstep);
  }

  for (int kt = 0; kt < NK; ++kt) {
    const int buf = kt & 1;

    wait_async0();
    __syncthreads();

    if (kt + 1 < NK) {
      const size_t k1 = (size_t)(kt + 1) * 128;
      const uint32_t dlds = (uint32_t)((buf ^ 1) * (128 * 128));
#pragma unroll
      for (int i = 0; i < 4; ++i) {
        const size_t rstep = (size_t)(32 * i) * DDIM;
        async_load16(ldsA0 + dlds + 32 * i * 128, Ag + k1 + rstep);
        async_load16(ldsB0 + dlds + 32 * i * 128, Bg + k1 + rstep);
      }
    }

#pragma unroll
    for (int ks = 0; ks < 2; ++ks) {
      v8i afrag[4], bfrag[2];
#pragma unroll
      for (int mt = 0; mt < 4; ++mt) {
        const int8_t* Arow =
            &As[buf][(wm * 64 + mt * 16 + ln) * 128 + ks * 64 + h * 8];
        int2 d0 = *(const int2*)(Arow + 0);
        int2 d1 = *(const int2*)(Arow + 16);
        int2 d2 = *(const int2*)(Arow + 32);
        int2 d3 = *(const int2*)(Arow + 48);
        afrag[mt][0] = d0.x; afrag[mt][1] = d0.y;
        afrag[mt][2] = d1.x; afrag[mt][3] = d1.y;
        afrag[mt][4] = d2.x; afrag[mt][5] = d2.y;
        afrag[mt][6] = d3.x; afrag[mt][7] = d3.y;
      }
#pragma unroll
      for (int nt = 0; nt < 2; ++nt) {
        const int8_t* Brow =
            &Bs[buf][(wn * 32 + nt * 16 + ln) * 128 + ks * 64 + h * 8];
        int2 e0 = *(const int2*)(Brow + 0);
        int2 e1 = *(const int2*)(Brow + 16);
        int2 e2 = *(const int2*)(Brow + 32);
        int2 e3 = *(const int2*)(Brow + 48);
        int4 lo = make_int4(e0.x, e0.y, e1.x, e1.y);
        int4 hi = make_int4(e2.x, e2.y, e3.x, e3.y);
        bfrag[nt][0] = lo.x; bfrag[nt][1] = lo.y;
        bfrag[nt][2] = lo.z; bfrag[nt][3] = lo.w;
        bfrag[nt][4] = hi.x; bfrag[nt][5] = hi.y;
        bfrag[nt][6] = hi.z; bfrag[nt][7] = hi.w;
      }
#pragma unroll
      for (int mt = 0; mt < 4; ++mt)
#pragma unroll
        for (int nt = 0; nt < 2; ++nt)
          acc[mt][nt] = __builtin_amdgcn_wmma_i32_16x16x64_iu8(
              true, afrag[mt], true, bfrag[nt], acc[mt][nt], false, false);
    }
  }

#pragma unroll
  for (int mt = 0; mt < 4; ++mt) {
#pragma unroll
    for (int nt = 0; nt < 2; ++nt) {
      const size_t colg = (size_t)(bn * 128 + wn * 32 + nt * 16 + ln);
#pragma unroll
      for (int g = 0; g < 8; ++g) {
        const int m = wm * 64 + mt * 16 + h * 8 + g;
        const float v = (float)acc[mt][nt][g] * xs[m] * gamma;
        out[(size_t)(bm * 128 + m) * ODIM + colg] = v;
      }
    }
  }
}

extern "C" void kernel_launch(void* const* d_in, const int* in_sizes, int n_in,
                              void* d_out, int out_size, void* d_ws, size_t ws_size,
                              hipStream_t stream) {
  const float* x = (const float*)d_in[0];
  const float* w = (const float*)d_in[1];
  float* out = (float*)d_out;

  char* ws = (char*)d_ws;
  int8_t* xq     = (int8_t*)ws;
  int8_t* wqbuf  = (int8_t*)(ws + (size_t)MTOK * DDIM);
  float*  xscale = (float*)(ws + (size_t)MTOK * DDIM + (size_t)ODIM * DDIM);
  double* absum  = (double*)(ws + (size_t)MTOK * DDIM + (size_t)ODIM * DDIM
                                + (size_t)MTOK * sizeof(float));

  quant_x_kernel<<<MTOK, 256, 0, stream>>>(x, xq, xscale);
  init_sum_kernel<<<1, 64, 0, stream>>>(absum);
  wabs_sum_kernel<<<8192, 256, 0, stream>>>(w, absum);
  quant_w_kernel<<<ODIM, 256, 0, stream>>>(w, absum, wqbuf);

  dim3 grid(ODIM / 128, MTOK / 128);
  gemm_iu8_kernel<<<grid, 256, 0, stream>>>(xq, wqbuf, xscale, absum, out);
}
